// BitwiseWavenetBlock_35785667510799
// MI455X (gfx1250) — matrix operand order repaired and checked by reading, not compiled, not run on hardware
//
#include <hip/hip_runtime.h>

typedef _Float16 v4h  __attribute__((ext_vector_type(4)));
typedef _Float16 v8h  __attribute__((ext_vector_type(8)));
typedef _Float16 v16h __attribute__((ext_vector_type(16)));
typedef float    v8f  __attribute__((ext_vector_type(8)));

#define LAYERS 10
#define CCH    128      // RES == DL == 128
#define BATCH  16
#define SEQ    16384
#define TILE_T 64

// per-layer packed-weight block (halves): WF[2][128][128] | WG[2][128][128] | WR[128][128] | WS[128][128]
#define WF_OFF 0
#define WG_OFF 32768
#define WR_OFF 65536
#define WS_OFF 81920
#define LAYER_W 98304

// ---------------------------------------------------------------------------
// init: resid0 = x, skip = 0
// ---------------------------------------------------------------------------
__global__ void init_out(const float* __restrict__ x, float* __restrict__ resid0,
                         float* __restrict__ skip, int n4) {
    int i = blockIdx.x * blockDim.x + threadIdx.x;
    if (i >= n4) return;
    float4 v = ((const float4*)x)[i];
    ((float4*)resid0)[i] = v;
    ((float4*)skip)[i] = make_float4(0.f, 0.f, 0.f, 0.f);
}

// ---------------------------------------------------------------------------
// fold sigmoid gate + channel scale into f16 weights, repacked [tap][o][i]
// ---------------------------------------------------------------------------
__global__ void prep_weights(const float* __restrict__ Wf, const float* __restrict__ Gf, const float* __restrict__ sf,
                             const float* __restrict__ Wg, const float* __restrict__ Gg, const float* __restrict__ sg,
                             const float* __restrict__ Wr, const float* __restrict__ Gr, const float* __restrict__ sr,
                             const float* __restrict__ Ws, const float* __restrict__ Gs, const float* __restrict__ ss,
                             _Float16* __restrict__ wEff) {
    int gid = blockIdx.x * blockDim.x + threadIdx.x;
    if (gid >= LAYERS * LAYER_W) return;
    int l = gid / LAYER_W;
    int r = gid % LAYER_W;
    float w, g, s;
    if (r < WG_OFF) {                                  // filter conv, K=2
        int tap = r >> 14, o = (r & 16383) >> 7, i = r & 127;
        int src = ((l * CCH + o) * CCH + i) * 2 + tap;
        w = Wf[src]; g = Gf[src]; s = sf[l * CCH + o];
    } else if (r < WR_OFF) {                           // gate conv, K=2
        int r2 = r - WG_OFF;
        int tap = r2 >> 14, o = (r2 & 16383) >> 7, i = r2 & 127;
        int src = ((l * CCH + o) * CCH + i) * 2 + tap;
        w = Wg[src]; g = Gg[src]; s = sg[l * CCH + o];
    } else if (r < WS_OFF) {                           // residual conv, K=1
        int r2 = r - WR_OFF;
        int o = r2 >> 7, i = r2 & 127;
        int src = (l * CCH + o) * CCH + i;
        w = Wr[src]; g = Gr[src]; s = sr[l * CCH + o];
    } else {                                           // skip conv, K=1
        int r2 = r - WS_OFF;
        int o = r2 >> 7, i = r2 & 127;
        int src = (l * CCH + o) * CCH + i;
        w = Ws[src]; g = Gs[src]; s = ss[l * CCH + o];
    }
    float sig = 1.0f / (1.0f + __expf(-g));
    wEff[gid] = (_Float16)(w * sig * s);
}

__global__ void prep_bias(const float* __restrict__ bf, const float* __restrict__ sf,
                          const float* __restrict__ bg, const float* __restrict__ sg,
                          const float* __restrict__ br, const float* __restrict__ sr,
                          const float* __restrict__ bs, const float* __restrict__ ss,
                          float* __restrict__ bEff) {
    int gid = blockIdx.x * blockDim.x + threadIdx.x;
    if (gid >= LAYERS * 4 * CCH) return;
    int l = gid / (4 * CCH);
    int r = gid % (4 * CCH);
    int which = r >> 7, o = r & 127;
    int src = l * CCH + o;
    float v;
    if      (which == 0) v = sf[src] * bf[src];
    else if (which == 1) v = sg[src] * bg[src];
    else if (which == 2) v = sr[src] * br[src];
    else                 v = ss[src] * bs[src];
    bEff[gid] = v;
}

// ---------------------------------------------------------------------------
// one WaveNet layer:  filtered/gated (K=2 dilated) -> product -> skip/resid (K=1)
// block = 8 wave32s; block owns (batch b, 64-time tile) x 128 out channels
// ---------------------------------------------------------------------------
__global__ __launch_bounds__(256)
void wavenet_layer(const float* __restrict__ residIn,
                   float*       __restrict__ residOut,
                   float*       __restrict__ skip,
                   const _Float16* __restrict__ wEffL,  // this layer's packed weights
                   const float*    __restrict__ bEffL,  // this layer's 4*128 biases
                   int tlOff, int trOff) {
    __shared__ __align__(32) _Float16 ldsX[2][TILE_T * CCH];   // [tap tile][t*128 + c]

    const int tid  = threadIdx.x;
    const int lane = tid & 31;          // wave32
    const int wv   = tid >> 5;          // 0..7
    const int hi   = lane >> 4;         // lane half (K-split in WMMA layouts)
    const int b    = blockIdx.y;
    const int t0   = blockIdx.x * TILE_T;

    // ---- stage 0: stage both shifted activation tiles into LDS as f16 [t][c] ----
    // mapping: one thread = one time column x 4 consecutive channels
    //   -> global reads coalesced along t, LDS write is a single ds_store_b64
    for (int tile = 0; tile < 2; ++tile) {
        const int off = tile ? trOff : tlOff;
        for (int it = 0; it < 8; ++it) {
            int vid = it * 256 + tid;            // 0..2047
            int t   = vid & 63;                  // lanes -> consecutive t
            int cg  = vid >> 6;                  // channel group: 4 channels each
            long tg = (long)t0 + off + t;
            const float* src = residIn + ((size_t)b * CCH + cg * 4) * SEQ + tg;
            float f0 = 0.f, f1 = 0.f, f2 = 0.f, f3 = 0.f;
            if ((unsigned long)tg < (unsigned long)SEQ) {
                f0 = src[0 * (size_t)SEQ];
                f1 = src[1 * (size_t)SEQ];
                f2 = src[2 * (size_t)SEQ];
                f3 = src[3 * (size_t)SEQ];
            }
            v4h pk = { (_Float16)f0, (_Float16)f1, (_Float16)f2, (_Float16)f3 };
            *(v4h*)&ldsX[tile][t * CCH + cg * 4] = pk;     // ds_store_b64
        }
    }
    __syncthreads();

    const int rowBase = wv * 16 + (lane & 15);   // A-matrix row this lane feeds

    // per-row folded biases (row depends on acc VGPR index r and lane half)
    float bF[8], bG[8], bR[8], bS[8];
    #pragma unroll
    for (int r = 0; r < 8; ++r) {
        int m = wv * 16 + r + hi * 8;
        bF[r] = bEffL[0 * CCH + m];
        bG[r] = bEffL[1 * CCH + m];
        bR[r] = bEffL[2 * CCH + m];
        bS[r] = bEffL[3 * CCH + m];
    }

    const v8f zero = {};
    v8f accF[4], accG[4];

    // ---- stage 1: filtered & gated GEMMs, K = 2 taps x 128 channels ----
    const _Float16* WF = wEffL + WF_OFF;
    const _Float16* WG = wEffL + WG_OFF;
    for (int nt = 0; nt < 4; ++nt) {
        v8f af = zero, ag = zero;
        const int nCol = nt * 16 + (lane & 15);
        for (int tap = 0; tap < 2; ++tap) {
            const _Float16* ldsB = ldsX[tap];
            #pragma unroll
            for (int kb = 0; kb < 4; ++kb) {
                const int kA = kb * 32 + hi * 8;     // ISA A layout: two 8-half runs / lane
                const _Float16* pF = WF + tap * 16384 + rowBase * CCH + kA;
                const _Float16* pG = WG + tap * 16384 + rowBase * CCH + kA;
                v8h fl = *(const v8h*)pF, fh = *(const v8h*)(pF + 16);
                v8h gl = *(const v8h*)pG, gh = *(const v8h*)(pG + 16);
                v16h aF, aG;
                #pragma unroll
                for (int j = 0; j < 8; ++j) {
                    aF[j] = fl[j]; aF[8 + j] = fh[j];
                    aG[j] = gl[j]; aG[8 + j] = gh[j];
                }
                // ISA B layout: two 8-half runs / lane (K 8h..8h+7, 16+8h..16+8h+7), matching A
                v16h bfrag = __builtin_shufflevector(*(const v8h*)&ldsB[nCol * CCH + kb * 32 + hi * 8], *(const v8h*)&ldsB[nCol * CCH + kb * 32 + 16 + hi * 8], 0, 1, 2, 3, 4, 5, 6, 7, 8, 9, 10, 11, 12, 13, 14, 15);
                af = __builtin_amdgcn_wmma_f32_16x16x32_f16(false, aF, false, bfrag, (short)0, af, false, false);
                ag = __builtin_amdgcn_wmma_f32_16x16x32_f16(false, aG, false, bfrag, (short)0, ag, false, false);
            }
        }
        accF[nt] = af; accG[nt] = ag;
    }
    __syncthreads();   // everyone done reading X tiles

    // ---- layer_out = (filtered+bf)*(gated+bg) -> LDS tile 0 as f16 [t][c] ----
    // rows m = wv*16 + hi*8 + r are contiguous for r = 0..7 -> one ds_store_b128
    for (int nt = 0; nt < 4; ++nt) {
        int n = nt * 16 + (lane & 15);
        v8h pk;
        #pragma unroll
        for (int r = 0; r < 8; ++r)
            pk[r] = (_Float16)((accF[nt][r] + bF[r]) * (accG[nt][r] + bG[r]));
        *(v8h*)&ldsX[0][n * CCH + wv * 16 + hi * 8] = pk;  // ds_store_b128
    }
    __syncthreads();

    // ---- stage 2: skip & residual 1x1 GEMMs, fused global accumulate ----
    const _Float16* WR = wEffL + WR_OFF;
    const _Float16* WS = wEffL + WS_OFF;
    for (int nt = 0; nt < 4; ++nt) {
        v8f ar = zero, as = zero;
        const int nCol = nt * 16 + (lane & 15);
        #pragma unroll
        for (int kb = 0; kb < 4; ++kb) {
            const int kA = kb * 32 + hi * 8;
            const _Float16* pR = WR + rowBase * CCH + kA;
            const _Float16* pS = WS + rowBase * CCH + kA;
            v8h rl = *(const v8h*)pR, rh = *(const v8h*)(pR + 16);
            v8h sl = *(const v8h*)pS, sh = *(const v8h*)(pS + 16);
            v16h aR, aS;
            #pragma unroll
            for (int j = 0; j < 8; ++j) {
                aR[j] = rl[j]; aR[8 + j] = rh[j];
                aS[j] = sl[j]; aS[8 + j] = sh[j];
            }
            v16h bfrag = __builtin_shufflevector(*(const v8h*)&ldsX[0][nCol * CCH + kb * 32 + hi * 8], *(const v8h*)&ldsX[0][nCol * CCH + kb * 32 + 16 + hi * 8], 0, 1, 2, 3, 4, 5, 6, 7, 8, 9, 10, 11, 12, 13, 14, 15);
            ar = __builtin_amdgcn_wmma_f32_16x16x32_f16(false, aR, false, bfrag, (short)0, ar, false, false);
            as = __builtin_amdgcn_wmma_f32_16x16x32_f16(false, aS, false, bfrag, (short)0, as, false, false);
        }
        #pragma unroll
        for (int r = 0; r < 8; ++r) {
            int m = wv * 16 + r + hi * 8;
            size_t gi = ((size_t)b * CCH + m) * SEQ + (size_t)t0 + nCol;
            residOut[gi] = residIn[gi] + ar[r] + bR[r];
            skip[gi]    = skip[gi]    + as[r] + bS[r];
        }
    }
}

// ---------------------------------------------------------------------------
extern "C" void kernel_launch(void* const* d_in, const int* in_sizes, int n_in,
                              void* d_out, int out_size, void* d_ws, size_t ws_size,
                              hipStream_t stream) {
    const float* x  = (const float*)d_in[0];
    const float* Wf = (const float*)d_in[1];
    const float* Gf = (const float*)d_in[2];
    const float* bf = (const float*)d_in[3];
    const float* sf = (const float*)d_in[4];
    const float* Wg = (const float*)d_in[5];
    const float* Gg = (const float*)d_in[6];
    const float* bg = (const float*)d_in[7];
    const float* sg = (const float*)d_in[8];
    const float* Wr = (const float*)d_in[9];
    const float* Gr = (const float*)d_in[10];
    const float* br = (const float*)d_in[11];
    const float* sr = (const float*)d_in[12];
    const float* Ws = (const float*)d_in[13];
    const float* Gs = (const float*)d_in[14];
    const float* bs = (const float*)d_in[15];
    const float* ss = (const float*)d_in[16];

    const size_t N = (size_t)BATCH * CCH * SEQ;          // 33.5M floats
    float* residA = (float*)d_out;                        // ping buffer 0 (final resid lands here)
    float* skip   = (float*)d_out + N;
    float* residB = (float*)d_ws;                         // ping buffer 1
    _Float16* wEff = (_Float16*)((char*)d_ws + N * sizeof(float));
    float*    bEff = (float*)(wEff + (size_t)LAYERS * LAYER_W);

    int n4 = (int)(N / 4);
    init_out<<<(n4 + 255) / 256, 256, 0, stream>>>(x, residA, skip, n4);

    int wtot = LAYERS * LAYER_W;
    prep_weights<<<(wtot + 255) / 256, 256, 0, stream>>>(Wf, Gf, sf, Wg, Gg, sg,
                                                         Wr, Gr, sr, Ws, Gs, ss, wEff);
    int btot = LAYERS * 4 * CCH;
    prep_bias<<<(btot + 255) / 256, 256, 0, stream>>>(bf, sf, bg, sg, br, sr, bs, ss, bEff);

    dim3 grid(SEQ / TILE_T, BATCH);
    for (int l = 0; l < LAYERS; ++l) {
        int d  = 1 << l;
        int tl = (l == 0) ? -1 : -(d >> 1);
        int tr = (l == 0) ?  0 :  (d >> 1);
        const float* rin  = (l & 1) ? residB : residA;
        float*       rout = (l & 1) ? residA : residB;
        wavenet_layer<<<grid, 256, 0, stream>>>(rin, rout, skip,
                                                wEff + (size_t)l * LAYER_W,
                                                bEff + l * 4 * CCH, tl, tr);
    }
    // layer 9 (odd) writes residA == d_out, so final resid is already in place.
}
